// LSTMModelClassify_5901285064985
// MI455X (gfx1250) — matrix operand order repaired and checked by reading, not compiled, not run on hardware
//
#include <hip/hip_runtime.h>
#include <hip/hip_bf16.h>

// 2-layer LSTM (B=2048, T=512, D=1, H=50) + FC(50->2), fully fused.
// One wave32 owns 16 batch rows; h and c state live in LDS (c in registers
// caused scratch spills -> reverted). Recurrent GEMMs on
// v_wmma_f32_16x16x32_bf16 with x/bias folded into augmented operands:
//   layer0: gates = [h0 | x_t | 1] @ [Whh0 | Wih0 | b0]^T      (K=52 -> 64)
//   layer1: gates = [h0' | h1 | 1] @ [Wih1 | Whh1 | b1]^T      (K=101 -> 128)
// Pointwise uses a static lane->(row, j-range) mapping: base pointers +
// strength-reduced immediate DS offsets, no div/mod, rolled loops (no spill),
// native v_tanh_f32 when available.

typedef __bf16 v16bf __attribute__((ext_vector_type(16)));
typedef float  v8f   __attribute__((ext_vector_type(8)));

#define T_LEN   512
#define NTILES  13              // 4H=200 padded to 13*16=208
#define WPB     2               // waves per block
#define FRAG_HALVES (78*512)    // 26 layer0 chunks + 52 layer1 chunks

__device__ __forceinline__ void lds_fence() {
  asm volatile("s_wait_dscnt 0" ::: "memory");
}

#if __has_builtin(__builtin_amdgcn_tanhf)
__device__ __forceinline__ float tanh_f(float x) { return __builtin_amdgcn_tanhf(x); }
__device__ __forceinline__ float sigm(float x)   { return 0.5f + 0.5f * __builtin_amdgcn_tanhf(0.5f * x); }
#elif __has_builtin(__builtin_amdgcn_tanh_f32)
__device__ __forceinline__ float tanh_f(float x) { return __builtin_amdgcn_tanh_f32(x); }
__device__ __forceinline__ float sigm(float x)   { return 0.5f + 0.5f * __builtin_amdgcn_tanh_f32(0.5f * x); }
#else
__device__ __forceinline__ float tanh_f(float x) { return 2.0f / (1.0f + __expf(-2.0f * x)) - 1.0f; }
__device__ __forceinline__ float sigm(float x)   { return 1.0f / (1.0f + __expf(-x)); }
#endif

// A-matrix (16x32 bf16) fragment from row-major LDS [16][stride] bf16.
// lanes 0-15 -> M=lane, K = kbase + {0..7, 16..23};
// lanes 16-31 -> M=lane-16, K = kbase + {8..15, 24..31}.
__device__ __forceinline__ v16bf load_a(const __bf16* mat, int stride, int lane, int kbase) {
  int M   = lane & 15;
  int off = (lane >> 4) << 3;
  const __bf16* p = mat + M * stride + kbase + off;
  v16bf a;
  ((uint4*)&a)[0] = *(const uint4*)(p);
  ((uint4*)&a)[1] = *(const uint4*)(p + 16);
  return a;
}

// B-matrix (32x16 bf16) fragment, pre-swizzled: chunk[lane*16 + e].
__device__ __forceinline__ v16bf load_b(const __bf16* chunk, int lane) {
  const __bf16* p = chunk + lane * 16;
  v16bf b;
  ((uint4*)&b)[0] = ((const uint4*)p)[0];
  ((uint4*)&b)[1] = ((const uint4*)p)[1];
  return b;
}

// Store v8f accumulator into transposed gates buffer gatesT[208][16]:
// VGPR r holds (M = Mb+r, N = Nc) -> consecutive addresses -> 2x ds_store_b128.
__device__ __forceinline__ void store_acc(float* gatesT, int lane, int nt, const v8f& acc) {
  int Nc = nt * 16 + (lane & 15);
  int Mb = (lane >> 4) * 8;
  float* gp = gatesT + Nc * 16 + Mb;
  ((uint4*)gp)[0] = ((const uint4*)&acc)[0];
  ((uint4*)gp)[1] = ((const uint4*)&acc)[1];
}

__global__ void lstm2_fused_kernel(const float* __restrict__ x,
                                   const float* __restrict__ Wih0,
                                   const float* __restrict__ Whh0,
                                   const float* __restrict__ bih0,
                                   const float* __restrict__ bhh0,
                                   const float* __restrict__ Wih1,
                                   const float* __restrict__ Whh1,
                                   const float* __restrict__ bih1,
                                   const float* __restrict__ bhh1,
                                   const float* __restrict__ Wfc,
                                   const float* __restrict__ bfc,
                                   float* __restrict__ out) {
  extern __shared__ char smem[];
  const int tid  = threadIdx.x;
  const int lane = tid & 31;
  const int wid  = tid >> 5;

  // ---- LDS layout ----
  __bf16* frags = (__bf16*)smem;                             // 79872 B
  char*   wbase = smem + FRAG_HALVES * 2;
  const int PW = 13312 + 2048 + 4096 + 1024 + 3200 + 3200;   // 26880 B per wave
  char*   my    = wbase + wid * PW;
  float*  gatesT = (float*)(my);                             // [208][16] f32 (transposed)
  __bf16* h0p    = (__bf16*)(my + 13312);                    // [16][64]  bf16: h0 | x | 1 | 0pad
  __bf16* hcat   = (__bf16*)(my + 13312 + 2048);             // [16][128] bf16: h0'|h1 | 1 | 0pad
  float*  xst    = (float*)(my + 13312 + 2048 + 4096);       // [16][16] staged x chunk
  float*  c0l    = xst + 256;                                // [16*50] cell state L0
  float*  c1l    = c0l + 800;                                // [16*50] cell state L1

  // ---- one-time: swizzle augmented weights (f32 -> bf16) into B-frag order ----
  // layer0 B[k][n]: k<50 -> Whh0[n][k]; k==50 -> Wih0[n]; k==51 -> bih0+bhh0; else 0.
  // layer1 B[k][n]: k<50 -> Wih1[n][k]; k<100 -> Whh1[n][k-50]; k==100 -> bih1+bhh1; else 0.
  for (int idx = tid; idx < FRAG_HALVES; idx += blockDim.x) {
    int chunk  = idx >> 9;
    int within = idx & 511;
    int ln = within >> 4;
    int h  = within & 15;
    int layer, nt, ks;
    if (chunk < 26) { layer = 0; nt = chunk >> 1; ks = chunk & 1; }
    else            { int c = chunk - 26; layer = 1; nt = c >> 2; ks = c & 3; }
    int n = nt * 16 + (ln & 15);
    int k = ks * 32 + ((ln >= 16) ? 8 : 0) + ((h < 8) ? h : 8 + h);
    float w = 0.f;
    if (n < 200) {
      if (layer == 0) {
        if (k < 50)       w = Whh0[n * 50 + k];
        else if (k == 50) w = Wih0[n];                 // x column (D=1)
        else if (k == 51) w = bih0[n] + bhh0[n];       // bias column
      } else {
        if (k < 50)        w = Wih1[n * 50 + k];
        else if (k < 100)  w = Whh1[n * 50 + (k - 50)];
        else if (k == 100) w = bih1[n] + bhh1[n];      // bias column
      }
    }
    frags[idx] = (__bf16)w;
  }
  // zero state, then plant the constant-1 columns (never overwritten)
  for (int i = lane; i < 16 * 64;  i += 32) h0p[i]  = (__bf16)0.f;
  for (int i = lane; i < 16 * 128; i += 32) hcat[i] = (__bf16)0.f;
  for (int i = lane; i < 800;      i += 32) { c0l[i] = 0.f; c1l[i] = 0.f; }
  if (lane < 16) {
    h0p[lane * 64 + 51]    = (__bf16)1.0f;
    hcat[lane * 128 + 100] = (__bf16)1.0f;
  }
  __syncthreads();

  const int rowbase = blockIdx.x * (WPB * 16) + wid * 16;

  // ---- static pointwise mapping: lane -> (row pb, j-range pj..pj+24) ----
  const int pb = lane & 15;            // batch row within tile
  const int pj = (lane >> 4) * 25;     // j base: 0 or 25
  const float*  gl  = gatesT + pj * 16 + pb;   // shared gate base (stride 16/elem)
  float*        cl0 = c0l + pb * 50 + pj;      // cell state base, layer 0
  float*        cl1 = c1l + pb * 50 + pj;      // cell state base, layer 1
  __bf16*       hl0 = h0p  + pb * 64 + pj;     // h0 write base
  __bf16*       hlc = hcat + pb * 128 + pj;    // hcat write base (layer0 region)
  __bf16*       hl1 = hcat + pb * 128 + 50 + pj; // hcat write base (layer1 region)

  // ---- x chunk prefetch: each lane owns (row = lane&15, t-half = lane>>4) ----
  const int xhalf = lane >> 4;         // 0 or 1: t offsets 0..7 / 8..15
  const float* xbase = x + (rowbase + pb) * T_LEN + xhalf * 8;
  float4 xf0 = ((const float4*)xbase)[0];     // chunk 0 in flight
  float4 xf1 = ((const float4*)(xbase + 4))[0];

  for (int t0 = 0; t0 < T_LEN; t0 += 16) {
    // commit staged chunk (waits on loads issued 16 steps ago)
    *(float4*)&xst[pb * 16 + xhalf * 8]     = xf0;
    *(float4*)&xst[pb * 16 + xhalf * 8 + 4] = xf1;
    lds_fence();
    // issue next chunk's loads; consumed at the next boundary
    if (t0 + 16 < T_LEN) {
      const float* np = xbase + (t0 + 16);
      xf0 = ((const float4*)np)[0];
      xf1 = ((const float4*)(np + 4))[0];
    }

    for (int tt = 0; tt < 16; ++tt) {
      // plant x_t into the augmented column of the layer-0 A operand
      if (lane < 16) h0p[lane * 64 + 50] = (__bf16)xst[lane * 16 + tt];
      lds_fence();

      // ---- layer 0: gates = [h0 | x | 1] @ [Whh0 | Wih0 | b0]^T ----
      {
        v16bf a0 = load_a(h0p, 64, lane, 0);
        v16bf a1 = load_a(h0p, 64, lane, 32);
        v16bf b0 = load_b(frags + 0 * 512, lane);
        v16bf b1 = load_b(frags + 1 * 512, lane);
        #pragma unroll
        for (int nt = 0; nt < NTILES; ++nt) {
          v16bf nb0, nb1;
          if (nt + 1 < NTILES) {               // prefetch next tile's B frags
            nb0 = load_b(frags + ((nt + 1) * 2 + 0) * 512, lane);
            nb1 = load_b(frags + ((nt + 1) * 2 + 1) * 512, lane);
          }
          v8f acc = {};
          acc = __builtin_amdgcn_wmma_f32_16x16x32_bf16(false, a0, false, b0, (short)0, acc, false, false);
          acc = __builtin_amdgcn_wmma_f32_16x16x32_bf16(false, a1, false, b1, (short)0, acc, false, false);
          store_acc(gatesT, lane, nt, acc);
          b0 = nb0; b1 = nb1;
        }
      }
      lds_fence();

      // ---- layer 0 pointwise (rolled; strength-reduced addressing) ----
      for (int q = 0; q < 25; ++q) {
        float gi = gl[(q)        * 16];
        float gf = gl[(50 + q)   * 16];
        float gg = gl[(100 + q)  * 16];
        float go = gl[(150 + q)  * 16];
        float cv = sigm(gf) * cl0[q] + sigm(gi) * tanh_f(gg);
        cl0[q] = cv;
        float hv = sigm(go) * tanh_f(cv);
        hl0[q] = (__bf16)hv;
        hlc[q] = (__bf16)hv;   // K 0..49 of the layer-1 concat operand
      }
      lds_fence();

      // ---- layer 1: gates = [h0' | h1 | 1] @ [Wih1 | Whh1 | b1]^T ----
      {
        v16bf a0 = load_a(hcat, 128, lane, 0);
        v16bf a1 = load_a(hcat, 128, lane, 32);
        v16bf a2 = load_a(hcat, 128, lane, 64);
        v16bf a3 = load_a(hcat, 128, lane, 96);
        const __bf16* cb0 = frags + 26 * 512;
        v16bf b0 = load_b(cb0 + 0 * 512, lane);
        v16bf b1 = load_b(cb0 + 1 * 512, lane);
        v16bf b2 = load_b(cb0 + 2 * 512, lane);
        v16bf b3 = load_b(cb0 + 3 * 512, lane);
        #pragma unroll
        for (int nt = 0; nt < NTILES; ++nt) {
          v16bf nb0, nb1, nb2, nb3;
          if (nt + 1 < NTILES) {               // prefetch next tile's B frags
            const __bf16* cb = frags + (26 + (nt + 1) * 4) * 512;
            nb0 = load_b(cb + 0 * 512, lane);
            nb1 = load_b(cb + 1 * 512, lane);
            nb2 = load_b(cb + 2 * 512, lane);
            nb3 = load_b(cb + 3 * 512, lane);
          }
          v8f acc = {};
          acc = __builtin_amdgcn_wmma_f32_16x16x32_bf16(false, a0, false, b0, (short)0, acc, false, false);
          acc = __builtin_amdgcn_wmma_f32_16x16x32_bf16(false, a1, false, b1, (short)0, acc, false, false);
          acc = __builtin_amdgcn_wmma_f32_16x16x32_bf16(false, a2, false, b2, (short)0, acc, false, false);
          acc = __builtin_amdgcn_wmma_f32_16x16x32_bf16(false, a3, false, b3, (short)0, acc, false, false);
          store_acc(gatesT, lane, nt, acc);
          b0 = nb0; b1 = nb1; b2 = nb2; b3 = nb3;
        }
      }
      lds_fence();

      // ---- layer 1 pointwise ----
      for (int q = 0; q < 25; ++q) {
        float gi = gl[(q)        * 16];
        float gf = gl[(50 + q)   * 16];
        float gg = gl[(100 + q)  * 16];
        float go = gl[(150 + q)  * 16];
        float cv = sigm(gf) * cl1[q] + sigm(gi) * tanh_f(gg);
        cl1[q] = cv;
        float hv = sigm(go) * tanh_f(cv);
        hl1[q] = (__bf16)hv;   // K 50..99 of the concat operand
      }
      lds_fence();
    }
  }

  // ---- FC on last hidden: out[b, cls] = h1[b,:] . Wfc[cls,:] + bfc[cls] ----
  {
    int b = lane >> 1, cls = lane & 1;
    float s = bfc[cls];
    for (int j = 0; j < 50; ++j)
      s += (float)hcat[b * 128 + 50 + j] * Wfc[cls * 50 + j];
    out[(rowbase + b) * 2 + cls] = s;
  }
}

extern "C" void kernel_launch(void* const* d_in, const int* in_sizes, int n_in,
                              void* d_out, int out_size, void* d_ws, size_t ws_size,
                              hipStream_t stream) {
  (void)in_sizes; (void)n_in; (void)out_size; (void)d_ws; (void)ws_size;
  const float* xx   = (const float*)d_in[0];
  const float* Wih0 = (const float*)d_in[1];
  const float* Whh0 = (const float*)d_in[2];
  const float* bih0 = (const float*)d_in[3];
  const float* bhh0 = (const float*)d_in[4];
  const float* Wih1 = (const float*)d_in[5];
  const float* Whh1 = (const float*)d_in[6];
  const float* bih1 = (const float*)d_in[7];
  const float* bhh1 = (const float*)d_in[8];
  const float* Wfc  = (const float*)d_in[9];
  const float* bfc  = (const float*)d_in[10];

  // LDS: 79872 (augmented weight frags) + 2*26880 (per-wave state)
  size_t smem = (size_t)(FRAG_HALVES * 2) + (size_t)WPB * 26880;
  dim3 grid(2048 / (WPB * 16));   // 64 blocks, each = 2 waves * 16 rows
  dim3 block(WPB * 32);
  lstm2_fused_kernel<<<grid, block, smem, stream>>>(
      xx, Wih0, Whh0, bih0, bhh0, Wih1, Whh1, bih1, bhh1, Wfc, bfc,
      (float*)d_out);
}
